// multiConv_71408126263435
// MI455X (gfx1250) — matrix operand order repaired and checked by reading, not compiled, not run on hardware
//
#include <hip/hip_runtime.h>
#include <hip/hip_bf16.h>

typedef __attribute__((ext_vector_type(16))) __bf16 v16bf;
typedef __attribute__((ext_vector_type(8)))  float  v8f;

#define B_SZ   2048
#define D_IN   1024
#define HID    2048
#define KSUB   32
#define M_TILE 128     // rows of x per workgroup
#define N_TILE 128     // hid columns per tile
#define KC     32      // d-chunk per WMMA pass
#define NCHUNK (D_IN / KC)   // 32
#define XS     40      // LDS row stride (halfwords): 80B rows, 16B aligned
#define WS     40

static __device__ __forceinline__ unsigned short bf16_bits(__bf16 h) {
    return __builtin_bit_cast(unsigned short, h);
}

__global__ __launch_bounds__(256) void fused_mlp_wmma(
    const float* __restrict__ x, const long long* __restrict__ subset1,
    const float* __restrict__ W1, const float* __restrict__ b1,
    const float* __restrict__ W2, const float* __restrict__ b2,
    float* __restrict__ out)
{
    // double-buffered stages: 2 * (10KB*4) = 80KB of the 320KB/WGP LDS
    __shared__ alignas(16) unsigned short sXhi[2][M_TILE * XS];
    __shared__ alignas(16) unsigned short sXlo[2][M_TILE * XS];
    __shared__ alignas(16) unsigned short sWhi[2][N_TILE * WS];
    __shared__ alignas(16) unsigned short sWlo[2][N_TILE * WS];

    const int tid  = threadIdx.x;
    const int lane = tid & 31;
    const int wave = tid >> 5;
    const int half = lane >> 4;   // 0/1
    const int l16  = lane & 15;

    const int bt   = blockIdx.x & 15;   // b-tile fast -> same-k blocks co-resident
    const int kidx = blockIdx.x >> 4;   // k slow -> W1[k] stays hot in L2
    const long long ksel = subset1[kidx];

    const float* __restrict__ W1k = W1 + (size_t)ksel * D_IN * HID;
    const float* __restrict__ b1k = b1 + (size_t)ksel * HID;
    const float* __restrict__ W2k = W2 + (size_t)ksel * HID;
    const float  b2k = b2[ksel];
    const int brow0 = bt * M_TILE;

    // X staging coords: idx = tid + jj*256 -> 128 rows x 8 float4 (coalesced)
    int xrow[4], xc4[4];
#pragma unroll
    for (int jj = 0; jj < 4; ++jj) {
        const int idx = tid + jj * 256;
        xrow[jj] = idx >> 3;  xc4[jj] = idx & 7;
    }
    // W staging coords: wave w owns d-rows 4w..4w+3; lanes sweep h (coalesced);
    // each thread converts a 4(d) x 4(h) block -> packed b64 transposed stores
    const int wdbase = (tid >> 5) * 4;   // d_local base (0..28)
    const int wh4    = tid & 31;         // h_local/4   (0..31)

    // A-fragment addressing (16-bit A 16x32): lanes 0-15 hold K 0..7 / 16..23,
    // lanes 16-31 hold K 8..15 / 24..31
    const int arow = wave * 16 + l16;
    const int kbA  = half * 8;
    // B-fragment: lane holds column l16, K = 8*half..+7 / 16+8*half..+7 (d-major)
    const int kbB  = half * 8;

    float pred_acc[8];
#pragma unroll
    for (int v = 0; v < 8; ++v) pred_acc[v] = 0.0f;

#pragma unroll 1
    for (int ht = 0; ht < HID; ht += N_TILE) {
        v8f acc[8];
#pragma unroll
        for (int j = 0; j < 8; ++j) acc[j] = (v8f){0,0,0,0,0,0,0,0};

        float4 xq[4];
        float  wqf[4][4];   // [d-row r][h-comp q]

        auto load_chunk = [&](int d0) {
#pragma unroll
            for (int jj = 0; jj < 4; ++jj)
                xq[jj] = *(const float4*)(x + (size_t)(brow0 + xrow[jj]) * D_IN + d0 + xc4[jj] * 4);
#pragma unroll
            for (int r = 0; r < 4; ++r) {
                const float4 v4 = *(const float4*)(W1k + (size_t)(d0 + wdbase + r) * HID + ht + wh4 * 4);
                wqf[r][0] = v4.x; wqf[r][1] = v4.y; wqf[r][2] = v4.z; wqf[r][3] = v4.w;
            }
        };

        auto stage_chunk = [&](int buf) {
            // X: row-major, packed b64 stores
#pragma unroll
            for (int jj = 0; jj < 4; ++jj) {
                const float4 v4 = xq[jj];
                const __bf16 h0 = (__bf16)v4.x, h1 = (__bf16)v4.y,
                             h2 = (__bf16)v4.z, h3 = (__bf16)v4.w;
                const __bf16 l0 = (__bf16)(v4.x - (float)h0), l1 = (__bf16)(v4.y - (float)h1),
                             l2 = (__bf16)(v4.z - (float)h2), l3 = (__bf16)(v4.w - (float)h3);
                ushort4 hv = { bf16_bits(h0), bf16_bits(h1), bf16_bits(h2), bf16_bits(h3) };
                ushort4 lv = { bf16_bits(l0), bf16_bits(l1), bf16_bits(l2), bf16_bits(l3) };
                *(ushort4*)(&sXhi[buf][xrow[jj] * XS + xc4[jj] * 4]) = hv;
                *(ushort4*)(&sXlo[buf][xrow[jj] * XS + xc4[jj] * 4]) = lv;
            }
            // W: transposed [h][d]; 4 consecutive d per h -> packed b64 stores
#pragma unroll
            for (int q = 0; q < 4; ++q) {
                const float f0 = wqf[0][q], f1 = wqf[1][q], f2 = wqf[2][q], f3 = wqf[3][q];
                const __bf16 h0 = (__bf16)f0, h1 = (__bf16)f1, h2 = (__bf16)f2, h3 = (__bf16)f3;
                const __bf16 l0 = (__bf16)(f0 - (float)h0), l1 = (__bf16)(f1 - (float)h1),
                             l2 = (__bf16)(f2 - (float)h2), l3 = (__bf16)(f3 - (float)h3);
                ushort4 hv = { bf16_bits(h0), bf16_bits(h1), bf16_bits(h2), bf16_bits(h3) };
                ushort4 lv = { bf16_bits(l0), bf16_bits(l1), bf16_bits(l2), bf16_bits(l3) };
                const int hl = wh4 * 4 + q;
                *(ushort4*)(&sWhi[buf][hl * WS + wdbase]) = hv;
                *(ushort4*)(&sWlo[buf][hl * WS + wdbase]) = lv;
            }
        };

        // prologue: chunk 0 into buffer 0
        load_chunk(0);
        stage_chunk(0);

#pragma unroll 1
        for (int it = 0; it < NCHUNK; ++it) {
            __syncthreads();            // buffer (it&1) staged & prior WAR cleared
            const int buf = it & 1;

            // issue next chunk's global loads early (overlap with WMMAs)
            if (it + 1 < NCHUNK) load_chunk((it + 1) * KC);

            // ---- A fragments for this wave ----
            v16bf a_hi, a_lo;
            ((uint4*)&a_hi)[0] = *(const uint4*)(&sXhi[buf][arow * XS + kbA]);
            ((uint4*)&a_hi)[1] = *(const uint4*)(&sXhi[buf][arow * XS + kbA + 16]);
            ((uint4*)&a_lo)[0] = *(const uint4*)(&sXlo[buf][arow * XS + kbA]);
            ((uint4*)&a_lo)[1] = *(const uint4*)(&sXlo[buf][arow * XS + kbA + 16]);

            // ---- 8 N-subtiles, 3 WMMAs each (hi*hi + hi*lo + lo*hi) ----
#pragma unroll
            for (int j = 0; j < 8; ++j) {
                const int jc = j * 16 + l16;
                v16bf b_hi, b_lo;
                ((uint4*)&b_hi)[0] = *(const uint4*)(&sWhi[buf][jc * WS + kbB]);
                ((uint4*)&b_hi)[1] = *(const uint4*)(&sWhi[buf][jc * WS + kbB + 16]);
                ((uint4*)&b_lo)[0] = *(const uint4*)(&sWlo[buf][jc * WS + kbB]);
                ((uint4*)&b_lo)[1] = *(const uint4*)(&sWlo[buf][jc * WS + kbB + 16]);

                acc[j] = __builtin_amdgcn_wmma_f32_16x16x32_bf16(
                    false, a_hi, false, b_hi, (short)0, acc[j], false, false);
                acc[j] = __builtin_amdgcn_wmma_f32_16x16x32_bf16(
                    false, a_hi, false, b_lo, (short)0, acc[j], false, false);
                acc[j] = __builtin_amdgcn_wmma_f32_16x16x32_bf16(
                    false, a_lo, false, b_hi, (short)0, acc[j], false, false);
            }

            // stage next chunk into the other buffer (no conflict with readers of `buf`)
            if (it + 1 < NCHUNK) stage_chunk(buf ^ 1);
        }

        // ---- epilogue: bias + relu, dot with W2, reduce over the 128 h columns ----
        // C/D layout: lane holds column n=l16 of subtile; VGPR v holds row m=v+8*half
        float s[8];
#pragma unroll
        for (int v = 0; v < 8; ++v) s[v] = 0.0f;
#pragma unroll
        for (int j = 0; j < 8; ++j) {
            const int c = ht + j * 16 + l16;
            const float bb = b1k[c];
            const float ww = W2k[c];
#pragma unroll
            for (int v = 0; v < 8; ++v) {
                float hval = acc[j][v] + bb;
                hval = fmaxf(hval, 0.0f);
                s[v] = fmaf(hval, ww, s[v]);
            }
        }
#pragma unroll
        for (int v = 0; v < 8; ++v) {
#pragma unroll
            for (int m = 1; m < 16; m <<= 1)
                s[v] += __shfl_xor(s[v], m, 32);
            pred_acc[v] += s[v];
        }
    }

    // ---- write pred[b, kidx]; rows for this thread: wave*16 + half*8 + v ----
    if (l16 == 0) {
#pragma unroll
        for (int v = 0; v < 8; ++v) {
            const int brow = brow0 + wave * 16 + half * 8 + v;
            out[(size_t)brow * KSUB + kidx] = pred_acc[v] + b2k;
        }
    }
}

extern "C" void kernel_launch(void* const* d_in, const int* in_sizes, int n_in,
                              void* d_out, int out_size, void* d_ws, size_t ws_size,
                              hipStream_t stream) {
    (void)in_sizes; (void)n_in; (void)d_ws; (void)ws_size; (void)out_size;
    const float*      x       = (const float*)d_in[0];
    const long long*  subset1 = (const long long*)d_in[1];
    const float*      W1      = (const float*)d_in[2];
    const float*      b1      = (const float*)d_in[3];
    const float*      W2      = (const float*)d_in[4];
    const float*      b2      = (const float*)d_in[5];
    float*            out     = (float*)d_out;

    // grid: 16 b-tiles (fast) x 32 k (slow) = 512 workgroups of 8 wave32s
    dim3 grid((B_SZ / M_TILE) * KSUB), block(256);
    hipLaunchKernelGGL(fused_mlp_wmma, grid, block, 0, stream,
                       x, subset1, W1, b1, W2, b2, out);
}
